// RotaryMultiheadAttention_884763262981
// MI455X (gfx1250) — matrix operand order repaired and checked by reading, not compiled, not run on hardware
//
#include <hip/hip_runtime.h>
#include <cstdint>

#define DEV __device__ __forceinline__

constexpr int Bsz  = 4;
constexpr int Lseq = 2048;
constexpr int Dm   = 2048;
constexpr int Hh   = 16;
constexpr int Dh   = 128;
constexpr int SEQ  = Bsz * Lseq;      // 8192
constexpr int NQKV = 3 * Dm;          // 6144
constexpr float SCALE = 0.08838834764831845f; // 1/sqrt(128)

typedef __attribute__((ext_vector_type(16))) __bf16 v16bf;
typedef __attribute__((ext_vector_type(8)))  float  v8f;

union Frag { v16bf v; uint4 q[2]; };

DEV uint16_t f2bf(float f) {
    union { float f; uint32_t u; } x; x.f = f;
    uint32_t r = x.u + 0x7FFFu + ((x.u >> 16) & 1u);   // round-to-nearest-even
    return (uint16_t)(r >> 16);
}
DEV float bf2f(uint16_t h) {
    union { uint32_t u; float f; } x; x.u = ((uint32_t)h) << 16; return x.f;
}

DEV v8f wmma_bf16(const Frag& a, const Frag& b, v8f c) {
    return __builtin_amdgcn_wmma_f32_16x16x32_bf16(
        false, a.v, false, b.v, (short)0, c, false, false);
}

// ---------------------------------------------------------------- converts
__global__ __launch_bounds__(256) void k_cvt(const float* __restrict__ src,
                                             uint16_t* __restrict__ dst, int n) {
    int i = blockIdx.x * 256 + threadIdx.x;
    if (i < n) dst[i] = f2bf(src[i]);
}

// src: K x N row-major f32 (K = Dm = 2048), dst: N x K bf16 (transposed)
__global__ __launch_bounds__(256) void k_cvtT(const float* __restrict__ src,
                                              uint16_t* __restrict__ dst, int N) {
    int i = blockIdx.x * 256 + threadIdx.x;      // over N*K
    int k = i & (Dm - 1);
    int n = i >> 11;
    if (n < N) dst[(size_t)n * Dm + k] = f2bf(src[(size_t)k * N + n]);
}

// ---------------------------------------------------------------- GEMM
// C(MxN) = A(MxK) * B(KxN) + bias.  A bf16 row-major, Bt bf16 = B transposed (NxK).
// K fixed at 2048. 256x128 tile per 256-thread block; each wave owns 64x64
// (16 C-frags): 16 ds_load_b128 feed 16 WMMAs per K-step.
template <bool OUTF32>
__global__ __launch_bounds__(256) void k_gemm(const uint16_t* __restrict__ A,
                                              const uint16_t* __restrict__ Bt,
                                              const float* __restrict__ bias,
                                              void* __restrict__ Cout,
                                              int M, int N) {
    constexpr int K   = Dm;
    constexpr int LDT = 40;                       // lds row stride (u16), 80 B
    __shared__ __align__(16) uint16_t lA[256 * LDT];   // 20 KB
    __shared__ __align__(16) uint16_t lB[128 * LDT];   // 10 KB

    const int tid  = threadIdx.x;
    const int m0   = blockIdx.x * 256;
    const int n0   = blockIdx.y * 128;
    const int w    = tid >> 5;
    const int wr   = w & 3;                       // 4 row-groups of 64
    const int wc   = w >> 2;                      // 2 col-groups of 64
    const int lane = tid & 31;
    const int nl   = lane & 15;
    const int half = lane >> 4;

    v8f acc[4][4];
    const v8f vzero = {0.f, 0.f, 0.f, 0.f, 0.f, 0.f, 0.f, 0.f};
#pragma unroll
    for (int i = 0; i < 4; ++i)
#pragma unroll
        for (int j = 0; j < 4; ++j) acc[i][j] = vzero;

    for (int k0 = 0; k0 < K; k0 += 32) {
        // stage tiles: A 256x32 (1024 x 16B chunks), B 128x32 (512 chunks)
#pragma unroll
        for (int r = 0; r < 4; ++r) {
            int c = tid + 256 * r;                // 0..1023
            int row = c >> 2, seg = c & 3;
            const uint4* ga = (const uint4*)(A + (size_t)(m0 + row) * K + k0);
            ((uint4*)(lA + row * LDT))[seg] = ga[seg];
        }
#pragma unroll
        for (int r = 0; r < 2; ++r) {
            int c = tid + 256 * r;                // 0..511
            int row = c >> 2, seg = c & 3;
            const uint4* gb = (const uint4*)(Bt + (size_t)(n0 + row) * K + k0);
            ((uint4*)(lB + row * LDT))[seg] = gb[seg];
        }
        if (k0 + 64 < K) {                        // prefetch two K-tiles ahead
            int row = tid >> 2, seg = tid & 3;
            __builtin_prefetch(A  + (size_t)(m0 + row) * K + (k0 + 64) + seg * 8, 0, 3);
            __builtin_prefetch(Bt + (size_t)(n0 + (row & 127)) * K + (k0 + 64) + seg * 8, 0, 3);
        }
        __syncthreads();

        Frag a[4], b[4];
#pragma unroll
        for (int i = 0; i < 4; ++i) {
            const uint4* pa = (const uint4*)(lA + (wr * 64 + i * 16 + nl) * LDT + half * 16);
            a[i].q[0] = pa[0];
            a[i].q[1] = pa[1];                    // +16 bytes
        }
#pragma unroll
        for (int j = 0; j < 4; ++j) {
            const uint4* pb = (const uint4*)(lB + (wc * 64 + j * 16 + nl) * LDT + half * 16);
            b[j].q[0] = pb[0];
            b[j].q[1] = pb[1];
        }
#pragma unroll
        for (int i = 0; i < 4; ++i)
#pragma unroll
            for (int j = 0; j < 4; ++j)
                acc[i][j] = wmma_bf16(a[i], b[j], acc[i][j]);
        __syncthreads();
    }

#pragma unroll
    for (int i = 0; i < 4; ++i) {
        const int rbase = m0 + wr * 64 + i * 16 + half * 8;
#pragma unroll
        for (int j = 0; j < 4; ++j) {
            int col  = n0 + wc * 64 + j * 16 + nl;
            float bv = bias[col];
#pragma unroll
            for (int e = 0; e < 8; ++e) {
                float val  = acc[i][j][e] + bv;
                size_t idx = (size_t)(rbase + e) * N + col;
                if (OUTF32) ((float*)Cout)[idx]    = val;
                else        ((uint16_t*)Cout)[idx] = f2bf(val);
            }
        }
    }
}

// ---------------------------------------------------------------- RoPE + V transpose
// qkv: (SEQ, 6144) bf16.  qr/kr: (B,H,L,Dh) bf16.  vt: (B,H,Dh,L) bf16.
__global__ __launch_bounds__(256) void k_rope(const uint16_t* __restrict__ qkv,
                                              uint16_t* __restrict__ qr,
                                              uint16_t* __restrict__ kr,
                                              uint16_t* __restrict__ vt) {
    int idx = blockIdx.x * 256 + threadIdx.x;     // B*H*L*64 threads
    int d2 = idx & 63;
    int l  = (idx >> 6) & (Lseq - 1);
    int h  = (idx >> 17) & (Hh - 1);
    int b  = idx >> 21;

    size_t row  = (size_t)(b * Lseq + l) * NQKV;
    int cbase   = h * Dh + 2 * d2;
    float q0 = bf2f(qkv[row + cbase]);
    float q1 = bf2f(qkv[row + cbase + 1]);
    float k0 = bf2f(qkv[row + Dm + cbase]);
    float k1 = bf2f(qkv[row + Dm + cbase + 1]);

    float freq = __powf(10000.f, -(float)(2 * d2) * (1.0f / 128.0f));
    float ang  = (float)l * freq;
    float s, c;
    __sincosf(ang, &s, &c);

    size_t o = ((size_t)(b * Hh + h) * Lseq + l) * Dh + d2;
    qr[o]      = f2bf(q0 * c + q1 * s);
    qr[o + 64] = f2bf(q1 * c - q0 * s);
    kr[o]      = f2bf(k0 * c + k1 * s);
    kr[o + 64] = f2bf(k1 * c - k0 * s);

    size_t vo = ((size_t)(b * Hh + h) * Dh + 2 * d2) * Lseq + l;
    vt[vo]        = qkv[row + 2 * Dm + cbase];
    vt[vo + Lseq] = qkv[row + 2 * Dm + cbase + 1];
}

// ---------------------------------------------------------------- flash attention
// One wave (32 threads) per 16-query block of one (b,h). K/V stream from L2.
__global__ __launch_bounds__(32) void k_attn(const uint16_t* __restrict__ qr,
                                             const uint16_t* __restrict__ kr,
                                             const uint16_t* __restrict__ vt,
                                             uint16_t* __restrict__ ao) {
    constexpr int LDT = 40;
    __shared__ __align__(16) uint16_t Pt[16 * LDT];  // 16x32 bf16 prob tile (padded)

    const int lane = threadIdx.x & 31;
    const int nl   = lane & 15;
    const int half = lane >> 4;
    const int qblk = blockIdx.x;                  // 0..127
    const int h    = blockIdx.y;
    const int b    = blockIdx.z;
    const size_t bh = (size_t)(b * Hh + h);

    // Q A-fragments: 16 rows x 128 (4 k-steps of 32)
    Frag aq[4];
    const uint16_t* qbase = qr + (bh * Lseq + (size_t)qblk * 16 + nl) * Dh;
#pragma unroll
    for (int kk = 0; kk < 4; ++kk) {
        const uint4* p = (const uint4*)(qbase + kk * 32 + half * 16);
        aq[kk].q[0] = p[0];
        aq[kk].q[1] = p[1];
    }

    v8f o[8];
    const v8f vzero = {0.f, 0.f, 0.f, 0.f, 0.f, 0.f, 0.f, 0.f};
#pragma unroll
    for (int j = 0; j < 8; ++j) o[j] = vzero;
    float mrun[8], lsum[8];
#pragma unroll
    for (int i = 0; i < 8; ++i) { mrun[i] = -1e30f; lsum[i] = 0.f; }

    for (int kc = 0; kc < Lseq; kc += 32) {
        // ---- load all K B-frags (two 16-key n-tiles x 4 k-steps), then WMMA burst
        const uint16_t* k0row = kr + (bh * Lseq + kc + nl) * Dh;
        const uint16_t* k1row = k0row + 16 * Dh;
        __builtin_prefetch(k0row + 32 * Dh, 0, 3);       // next key chunk
        __builtin_prefetch(k1row + 32 * Dh, 0, 3);
        Frag bk0[4], bk1[4];
#pragma unroll
        for (int kk = 0; kk < 4; ++kk) {
            const uint4* p0 = (const uint4*)(k0row + kk * 32 + half * 16);
            const uint4* p1 = (const uint4*)(k1row + kk * 32 + half * 16);
            bk0[kk].q[0] = p0[0]; bk0[kk].q[1] = p0[1];
            bk1[kk].q[0] = p1[0]; bk1[kk].q[1] = p1[1];
        }
        v8f s0 = vzero, s1 = vzero;
#pragma unroll
        for (int kk = 0; kk < 4; ++kk) {
            s0 = wmma_bf16(aq[kk], bk0[kk], s0);
            s1 = wmma_bf16(aq[kk], bk1[kk], s1);
        }

        // ---- online softmax (rows of C-frag live across 16 lanes of a half)
#pragma unroll
        for (int i = 0; i < 8; ++i) {
            float e0 = s0[i] * SCALE;
            float e1 = s1[i] * SCALE;
            float v = fmaxf(e0, e1);
#pragma unroll
            for (int mm = 1; mm < 16; mm <<= 1) v = fmaxf(v, __shfl_xor(v, mm, 16));
            float nm   = fmaxf(mrun[i], v);
            float corr = __expf(mrun[i] - nm);
            mrun[i] = nm;
            float p0 = __expf(e0 - nm);
            float p1 = __expf(e1 - nm);
            float rs = p0 + p1;
#pragma unroll
            for (int mm = 1; mm < 16; mm <<= 1) rs += __shfl_xor(rs, mm, 16);
            lsum[i] = lsum[i] * corr + rs;
#pragma unroll
            for (int j = 0; j < 8; ++j) o[j][i] *= corr;
            Pt[(i + half * 8) * LDT + nl]      = f2bf(p0);
            Pt[(i + half * 8) * LDT + 16 + nl] = f2bf(p1);
        }

        // ---- transpose P through LDS into an A-fragment (in-order LDS, same wave)
        Frag ap;
        const uint4* pp = (const uint4*)(Pt + nl * LDT + half * 16);
        ap.q[0] = pp[0];
        ap.q[1] = pp[1];

        // ---- O += P . V   (B-frags read straight from v-transposed global)
        const uint16_t* vbase = vt + (bh * Dh + nl) * Lseq + kc + half * 16;
        __builtin_prefetch(vbase + 32, 0, 3);            // next key chunk
        Frag bv[8];
#pragma unroll
        for (int j = 0; j < 8; ++j) {
            const uint4* pv = (const uint4*)(vbase + (size_t)j * 16 * Lseq);
            bv[j].q[0] = pv[0];
            bv[j].q[1] = pv[1];
        }
#pragma unroll
        for (int j = 0; j < 8; ++j) o[j] = wmma_bf16(ap, bv[j], o[j]);
    }

    // ---- normalize and store (b, l, h*Dh+d) row-major bf16
#pragma unroll
    for (int i = 0; i < 8; ++i) {
        float inv  = 1.0f / lsum[i];
        size_t row = (size_t)(b * Lseq + qblk * 16 + i + half * 8) * Dm;
#pragma unroll
        for (int j = 0; j < 8; ++j)
            ao[row + h * Dh + j * 16 + nl] = f2bf(o[j][i] * inv);
    }
}

// ---------------------------------------------------------------- launcher
extern "C" void kernel_launch(void* const* d_in, const int* in_sizes, int n_in,
                              void* d_out, int out_size, void* d_ws, size_t ws_size,
                              hipStream_t stream) {
    (void)in_sizes; (void)n_in; (void)out_size; (void)ws_size;
    const float* x    = (const float*)d_in[0];
    const float* Wqkv = (const float*)d_in[1];
    const float* bqkv = (const float*)d_in[2];
    const float* Wout = (const float*)d_in[3];
    const float* bout = (const float*)d_in[4];

    char* ws = (char*)d_ws;
    size_t off = 0;
    uint16_t* xb   = (uint16_t*)(ws + off); off += (size_t)SEQ * Dm * 2;     // 33.5 MB
    uint16_t* wqT  = (uint16_t*)(ws + off); off += (size_t)Dm * NQKV * 2;    // 25.2 MB
    uint16_t* woT  = (uint16_t*)(ws + off); off += (size_t)Dm * Dm * 2;      //  8.4 MB
    uint16_t* qkvb = (uint16_t*)(ws + off); off += (size_t)SEQ * NQKV * 2;   // 100.7 MB
    uint16_t* qrb  = (uint16_t*)(ws + off); off += (size_t)SEQ * Dm * 2;     // 33.5 MB
    uint16_t* krb  = (uint16_t*)(ws + off); off += (size_t)SEQ * Dm * 2;     // 33.5 MB
    uint16_t* vtb  = (uint16_t*)(ws + off); off += (size_t)SEQ * Dm * 2;     // 33.5 MB
    uint16_t* aob  = xb;   // x-bf16 is dead after the QKV GEMM; reuse for attn out

    // 1) bf16 conversions (+ weight transposes for b128 fragment loads)
    k_cvt <<<(SEQ * Dm + 255) / 256, 256, 0, stream>>>(x, xb, SEQ * Dm);
    k_cvtT<<<(Dm * NQKV + 255) / 256, 256, 0, stream>>>(Wqkv, wqT, NQKV);
    k_cvtT<<<(Dm * Dm + 255) / 256, 256, 0, stream>>>(Wout, woT, Dm);

    // 2) QKV projection (8192x6144, K=2048), bf16 out
    k_gemm<false><<<dim3(SEQ / 256, NQKV / 128), 256, 0, stream>>>(
        xb, wqT, bqkv, qkvb, SEQ, NQKV);

    // 3) RoPE + V transpose
    k_rope<<<(Bsz * Hh * Lseq * 64) / 256, 256, 0, stream>>>(qkvb, qrb, krb, vtb);

    // 4) flash attention, one wave per (b, h, 16-query block)
    k_attn<<<dim3(Lseq / 16, Hh, Bsz), 32, 0, stream>>>(qrb, krb, vtb, aob);

    // 5) output projection (8192x2048, K=2048), f32 out + bias
    k_gemm<true><<<dim3(SEQ / 256, Dm / 128), 256, 0, stream>>>(
        aob, woT, bout, d_out, SEQ, Dm);
}
